// Phi3Transformer_54357106098451
// MI455X (gfx1250) — matrix operand order repaired and checked by reading, not compiled, not run on hardware
//
#include <hip/hip_runtime.h>
#include <cstdint>

// ---------------------------------------------------------------------------
// Phi3-like transformer forward for gfx1250 (MI455X).
// bf16 WMMA (f32 accumulate); weights pre-transposed/converted to bf16 [N][K]
// once per GEMM (live in 192MB L2); all GEMM/attention tiles staged into LDS
// via global_load_async_to_lds_b128 (ASYNCcnt), double-buffered k-loop.
// ---------------------------------------------------------------------------

#define H    2048
#define SEQ  2048
#define NH   32
#define HD   64
#define IM   8192
#define NLAYERS 2
#define LN_EPS 1e-5f

typedef __bf16 bf16;
typedef __attribute__((ext_vector_type(16))) __bf16 v16bf;
typedef __attribute__((ext_vector_type(8)))  __bf16 v8bf;
typedef __attribute__((ext_vector_type(8)))  float  v8f;
typedef __attribute__((ext_vector_type(4)))  float  v4f;

// ---------------------------------------------------------------------------
// CDNA5 async copy global->LDS (16B per lane) + ASYNCcnt waits.
// Low 32 bits of a generic pointer to LDS == LDS byte offset (ISA 10.2).
// ---------------------------------------------------------------------------
static __device__ __forceinline__ void cp_async_b128(void* lds_ptr, const void* gptr) {
  unsigned lds = (unsigned)(uintptr_t)lds_ptr;
  asm volatile("global_load_async_to_lds_b128 %0, %1, off"
               :: "v"(lds), "v"(gptr) : "memory");
}
template <int N>
static __device__ __forceinline__ void async_wait() {
  asm volatile("s_wait_asynccnt %0" :: "n"(N) : "memory");
}

// ---------------------------------------------------------------------------
// WMMA fragment helpers (CDNA5 ISA 7.12.2, 16-bit 16x16x32):
//  A: lane L<16 -> row L, K={kb..kb+7, kb+16..kb+23}, kb=0; lanes>=16: kb=8
//  B: lane L -> col L&15, K={kb..kb+7, kb+16..kb+23}, kb=0/8 by lane half
// ---------------------------------------------------------------------------
static __device__ __forceinline__ v8f wmma_bf16(v16bf a, v16bf b, v8f c) {
  return __builtin_amdgcn_wmma_f32_16x16x32_bf16(false, a, false, b,
                                                 (short)0, c, false, false);
}

static __device__ __forceinline__ v16bf ld_a_frag(const bf16* sm, int stride,
                                                  int row0, int kt) {
  int lane = threadIdx.x & 31;
  int r  = row0 + (lane & 15);
  int kb = kt * 32 + ((lane < 16) ? 0 : 8);
  const bf16* p = sm + r * stride + kb;
  union { v16bf v; v8bf h[2]; } u;
  u.h[0] = *(const v8bf*)(p);
  u.h[1] = *(const v8bf*)(p + 16);
  return u.v;
}

static __device__ __forceinline__ v16bf ld_b_frag(const bf16* sm, int stride,
                                                  int col0, int kt) {
  int lane = threadIdx.x & 31;
  int n  = col0 + (lane & 15);
  int kb = kt * 32 + ((lane < 16) ? 0 : 8);
  const bf16* p = sm + n * stride + kb;
  union { v16bf v; v8bf h[2]; } u;
  u.h[0] = *(const v8bf*)(p);
  u.h[1] = *(const v8bf*)(p + 16);
  return u.v;
}

// Branch-free tanh-GELU: tanh(z) = 1 - 2/(e^{2z}+1); saturates via inf.
static __device__ __forceinline__ float gelu_tanh(float x) {
  float z = 0.7978845608028654f * x * (1.0f + 0.044715f * x * x);
  float t = 1.0f - 2.0f / (__expf(2.0f * z) + 1.0f);
  return 0.5f * x * (1.0f + t);
}

// ---------------------------------------------------------------------------
// Tiled transpose + convert to bf16:  Win [R][C] (f32 or bf16) -> Wt [C][R].
// ---------------------------------------------------------------------------
template <typename TIN>
__global__ __launch_bounds__(256)
void transp_kernel(const TIN* __restrict__ Win, bf16* __restrict__ Wt,
                   int R, int C) {
  __shared__ float tile[32][33];
  const int r0 = blockIdx.y * 32, c0 = blockIdx.x * 32;
  const int tx = threadIdx.x & 31, ty = threadIdx.x >> 5;
#pragma unroll
  for (int i = 0; i < 4; ++i)
    tile[ty + i * 8][tx] = (float)Win[(size_t)(r0 + ty + i * 8) * C + c0 + tx];
  __syncthreads();
#pragma unroll
  for (int i = 0; i < 4; ++i)
    Wt[(size_t)(c0 + ty + i * 8) * R + r0 + tx] = (bf16)tile[tx][ty + i * 8];
}

// ---------------------------------------------------------------------------
// LayerNorm (center=False semantics). OUT = bf16 (hidden) or float (final).
// ---------------------------------------------------------------------------
template <typename OUT>
__global__ __launch_bounds__(256)
void ln_kernel(const float* __restrict__ x, const float* __restrict__ g,
               OUT* __restrict__ out) {
  __shared__ float red[256];
  const int row = blockIdx.x;
  const int tid = threadIdx.x;
  const v4f* xr4 = (const v4f*)(x + (size_t)row * H);

  v4f va = xr4[tid], vb = xr4[tid + 256];
  float s = (va[0] + va[1]) + (va[2] + va[3]) + (vb[0] + vb[1]) + (vb[2] + vb[3]);
  red[tid] = s; __syncthreads();
  for (int o = 128; o > 0; o >>= 1) { if (tid < o) red[tid] += red[tid + o]; __syncthreads(); }
  const float mu = red[0] * (1.0f / H);
  __syncthreads();

  float vs = 0.f;
#pragma unroll
  for (int i = 0; i < 4; ++i) { float d0 = va[i] - mu, d1 = vb[i] - mu; vs += d0 * d0 + d1 * d1; }
  red[tid] = vs; __syncthreads();
  for (int o = 128; o > 0; o >>= 1) { if (tid < o) red[tid] += red[tid + o]; __syncthreads(); }
  const float rstd = rsqrtf(red[0] * (1.0f / H) + LN_EPS);

  const v4f* g4 = (const v4f*)g;
  v4f ga = g4[tid], gb = g4[tid + 256];
  OUT* orow = out + (size_t)row * H;
#pragma unroll
  for (int i = 0; i < 4; ++i) {
    orow[tid * 4 + i]         = (OUT)((va[i] - mu) * rstd * ga[i]);
    orow[(tid + 256) * 4 + i] = (OUT)((vb[i] - mu) * rstd * gb[i]);
  }
}

// ---------------------------------------------------------------------------
// GEMM: C[M,N] = epi(A[M,K] @ Bt[N,K]^T).  A, Bt bf16; async double-buffered
// LDS staging; 128x128x64 block tile; 8 waves (2M x 4N), wave = 64x32 = 4x2
// fragments, 2 WMMA k-steps per staged tile (16 WMMAs / barrier pair).
// ---------------------------------------------------------------------------
#define BM 128
#define BN 128
#define BK 64
#define LDA_S 72
#define LDB_S 72

static __device__ __forceinline__ void stage_tiles(bf16* Asb, bf16* Bsb,
                                                   const bf16* A, const bf16* Bt,
                                                   int bm, int bn, int k0, int K,
                                                   int tid) {
#pragma unroll
  for (int i = 0; i < 4; ++i) {           // 1024 A chunks of 16B
    int id = tid + i * 256;
    int r = id >> 3, c = id & 7;
    cp_async_b128(Asb + r * LDA_S + c * 8, A + (size_t)(bm + r) * K + k0 + c * 8);
  }
#pragma unroll
  for (int i = 0; i < 4; ++i) {           // 1024 B chunks of 16B
    int id = tid + i * 256;
    int n = id >> 3, c = id & 7;
    cp_async_b128(Bsb + n * LDB_S + c * 8, Bt + (size_t)(bn + n) * K + k0 + c * 8);
  }
}

template <int OUT_BF16, int BIAS_RES, int GELU>
__global__ __launch_bounds__(256)
void gemm_bf16_kernel(const bf16* __restrict__ A, const bf16* __restrict__ Bt,
                      void* Cout, const float* __restrict__ bias,
                      const float* resid, int M, int N, int K) {
  __shared__ __align__(16) bf16 As[2][BM * LDA_S];
  __shared__ __align__(16) bf16 Bs[2][BN * LDB_S];
  (void)M;

  const int tid  = threadIdx.x;
  const int lane = tid & 31;
  const int wave = tid >> 5;
  const int wm = wave & 1;
  const int wn = wave >> 1;
  const int bm = blockIdx.y * BM;
  const int bn = blockIdx.x * BN;

  v8f acc[4][2];
#pragma unroll
  for (int mf = 0; mf < 4; ++mf)
#pragma unroll
    for (int nf = 0; nf < 2; ++nf)
      acc[mf][nf] = (v8f){0.f, 0.f, 0.f, 0.f, 0.f, 0.f, 0.f, 0.f};

  const int nsteps = K / BK;
  stage_tiles(As[0], Bs[0], A, Bt, bm, bn, 0, K, tid);

  for (int s = 0; s < nsteps; ++s) {
    const int cur = s & 1;
    if (s + 1 < nsteps) {
      stage_tiles(As[cur ^ 1], Bs[cur ^ 1], A, Bt, bm, bn, (s + 1) * BK, K, tid);
      async_wait<8>();          // 8 copies for next tile may remain in flight
    } else {
      async_wait<0>();
    }
    __syncthreads();

#pragma unroll
    for (int kt = 0; kt < 2; ++kt) {
      v16bf af[4], bfr[2];
#pragma unroll
      for (int mf = 0; mf < 4; ++mf) af[mf]  = ld_a_frag(As[cur], LDA_S, wm * 64 + mf * 16, kt);
#pragma unroll
      for (int nf = 0; nf < 2; ++nf) bfr[nf] = ld_b_frag(Bs[cur], LDB_S, wn * 32 + nf * 16, kt);
#pragma unroll
      for (int mf = 0; mf < 4; ++mf)
#pragma unroll
        for (int nf = 0; nf < 2; ++nf)
          acc[mf][nf] = wmma_bf16(af[mf], bfr[nf], acc[mf][nf]);
    }
    __syncthreads();
  }

#pragma unroll
  for (int mf = 0; mf < 4; ++mf) {
#pragma unroll
    for (int nf = 0; nf < 2; ++nf) {
      v8f v = acc[mf][nf];
#pragma unroll
      for (int e = 0; e < 8; ++e) {
        int row = bm + wm * 64 + mf * 16 + e + ((lane >> 4) << 3);
        int col = bn + wn * 32 + nf * 16 + (lane & 15);
        float val = v[e];
        if constexpr (BIAS_RES) val += bias[col];
        if constexpr (GELU)     val = gelu_tanh(val);
        if constexpr (BIAS_RES) val += resid[(size_t)row * N + col];
        if constexpr (OUT_BF16) ((bf16*)Cout)[(size_t)row * N + col] = (bf16)val;
        else                    ((float*)Cout)[(size_t)row * N + col] = val;
      }
    }
  }
}

// ---------------------------------------------------------------------------
// Flash-style causal attention. grid = (NH, SEQ/128), 256 threads (8 waves).
// q,k bf16 [S][H]; vt bf16 [H][S] (pre-transposed V); ctx bf16 [S][H].
// Tiles async-copied into LDS. Scores f32 in LDS; softmax writes P as bf16
// (packed b32 stores) so P@V uses plain bf16 fragment loads.
// ---------------------------------------------------------------------------
#define BQ  128
#define BKV 128
#define QS_LD 72
#define KS_LD 72
#define VT_LD 136
#define SS_LD 132
#define PS_LD 136

__global__ __launch_bounds__(256)
void attn_kernel(const bf16* __restrict__ q, const bf16* __restrict__ k,
                 const bf16* __restrict__ vt, bf16* __restrict__ ctx) {
  __shared__ __align__(16) bf16  Qs[BQ * QS_LD];
  __shared__ __align__(16) bf16  Ks[BKV * KS_LD];
  __shared__ __align__(16) bf16  Vt[HD * VT_LD];   // [d][kv]
  __shared__ __align__(16) float Ss[BQ * SS_LD];   // raw scores (f32)
  __shared__ __align__(16) bf16  Ps[BQ * PS_LD];   // softmax'd P (bf16)
  __shared__ float mrow[BQ], lrow[BQ], rsc[BQ];

  const int head  = blockIdx.x;
  const int qb    = blockIdx.y;
  const int qbase = qb * BQ;
  const int tid   = threadIdx.x;
  const int lane  = tid & 31;
  const int wave  = tid >> 5;
  const int wm = wave & 1;
  const int wn = wave >> 1;
  const float scale = 0.125f;   // 1/sqrt(64)

  // Async-stage Q block (128 x 64 bf16 = 1024 x 16B chunks).
#pragma unroll
  for (int i = 0; i < 4; ++i) {
    int id = tid + i * 256;
    int r = id >> 3, c = id & 7;
    cp_async_b128(Qs + r * QS_LD + c * 8,
                  q + (size_t)(qbase + r) * H + head * HD + c * 8);
  }
  if (tid < BQ) { mrow[tid] = -1e30f; lrow[tid] = 0.f; }

  v8f cacc[4];
#pragma unroll
  for (int mf = 0; mf < 4; ++mf)
    cacc[mf] = (v8f){0.f, 0.f, 0.f, 0.f, 0.f, 0.f, 0.f, 0.f};

  for (int jb = 0; jb <= qb; ++jb) {
    const int jbase = jb * BKV;
    __syncthreads();            // prior consumers of Ks/Vt/Ps done

    // Async-stage K (row-major) and V^T (row-major in [d][s]).
#pragma unroll
    for (int i = 0; i < 4; ++i) {
      int id = tid + i * 256;
      int r = id >> 3, c = id & 7;
      cp_async_b128(Ks + r * KS_LD + c * 8,
                    k + (size_t)(jbase + r) * H + head * HD + c * 8);
    }
#pragma unroll
    for (int i = 0; i < 4; ++i) {
      int id = tid + i * 256;
      int d = id >> 4, c = id & 15;
      cp_async_b128(Vt + d * VT_LD + c * 8,
                    vt + (size_t)(head * HD + d) * SEQ + jbase + c * 8);
    }
    async_wait<0>();
    __syncthreads();

    // S = Q @ K^T (128x128, K=64).
    v8f sacc[4][2];
#pragma unroll
    for (int mf = 0; mf < 4; ++mf)
#pragma unroll
      for (int nf = 0; nf < 2; ++nf)
        sacc[mf][nf] = (v8f){0.f, 0.f, 0.f, 0.f, 0.f, 0.f, 0.f, 0.f};
#pragma unroll
    for (int kt = 0; kt < 2; ++kt) {
      v16bf af[4], bfr[2];
#pragma unroll
      for (int mf = 0; mf < 4; ++mf) af[mf]  = ld_a_frag(Qs, QS_LD, wm * 64 + mf * 16, kt);
#pragma unroll
      for (int nf = 0; nf < 2; ++nf) bfr[nf] = ld_b_frag(Ks, KS_LD, wn * 32 + nf * 16, kt);
#pragma unroll
      for (int mf = 0; mf < 4; ++mf)
#pragma unroll
        for (int nf = 0; nf < 2; ++nf)
          sacc[mf][nf] = wmma_bf16(af[mf], bfr[nf], sacc[mf][nf]);
    }
#pragma unroll
    for (int mf = 0; mf < 4; ++mf)
#pragma unroll
      for (int nf = 0; nf < 2; ++nf)
#pragma unroll
        for (int e = 0; e < 8; ++e) {
          int r = wm * 64 + mf * 16 + e + ((lane >> 4) << 3);
          int c = wn * 32 + nf * 16 + (lane & 15);
          Ss[r * SS_LD + c] = sacc[mf][nf][e] * scale;
        }
    __syncthreads();

    // Online softmax, one thread per row; emit P as packed bf16 pairs.
    if (tid < BQ) {
      const int r = tid, gr = qbase + r;
      const float* srow = &Ss[r * SS_LD];
      bf16* prow = &Ps[r * PS_LD];
      float mold = mrow[r], mx = mold;
      for (int c = 0; c < BKV; ++c) {
        float s2 = ((jbase + c) <= gr) ? srow[c] : -1e30f;
        mx = fmaxf(mx, s2);
      }
      float sum = 0.f;
      for (int c = 0; c < BKV; c += 2) {
        float s0 = ((jbase + c)     <= gr) ? srow[c]     : -1e30f;
        float s1 = ((jbase + c + 1) <= gr) ? srow[c + 1] : -1e30f;
        float p0 = __expf(s0 - mx), p1 = __expf(s1 - mx);
        union { bf16 h[2]; unsigned u; } pk;
        pk.h[0] = (bf16)p0; pk.h[1] = (bf16)p1;
        *(unsigned*)(prow + c) = pk.u;
        sum += p0 + p1;
      }
      rsc[r]  = __expf(mold - mx);
      mrow[r] = mx;
      lrow[r] = lrow[r] * rsc[r] + sum;
    }
    __syncthreads();

    // ctx = ctx * rsc + P @ V (each wave: 64 rows x 16 dims).
#pragma unroll
    for (int mf = 0; mf < 4; ++mf)
#pragma unroll
      for (int e = 0; e < 8; ++e) {
        int r = wm * 64 + mf * 16 + e + ((lane >> 4) << 3);
        cacc[mf][e] = cacc[mf][e] * rsc[r];
      }
#pragma unroll
    for (int kt = 0; kt < 4; ++kt) {
      // Batch all fragment loads before the WMMA chain so the ds_load_b128s
      // issue together and the s_wait_dscnt waits stagger across WMMAs.
      v16bf vb = ld_b_frag(Vt, VT_LD, wn * 16, kt);
      v16bf pa[4];
#pragma unroll
      for (int mf = 0; mf < 4; ++mf)
        pa[mf] = ld_a_frag(Ps, PS_LD, wm * 64 + mf * 16, kt);
#pragma unroll
      for (int mf = 0; mf < 4; ++mf)
        cacc[mf] = wmma_bf16(pa[mf], vb, cacc[mf]);
    }
  }

#pragma unroll
  for (int mf = 0; mf < 4; ++mf)
#pragma unroll
    for (int e = 0; e < 8; ++e) {
      int rl  = wm * 64 + mf * 16 + e + ((lane >> 4) << 3);
      int col = head * HD + wn * 16 + (lane & 15);
      ctx[(size_t)(qbase + rl) * H + col] = (bf16)(cacc[mf][e] / lrow[rl]);
    }
}

// ---------------------------------------------------------------------------
// Host orchestration.
// ---------------------------------------------------------------------------
extern "C" void kernel_launch(void* const* d_in, const int* in_sizes, int n_in,
                              void* d_out, int out_size, void* d_ws, size_t ws_size,
                              hipStream_t stream) {
  (void)in_sizes; (void)n_in; (void)out_size; (void)ws_size;
  const float* embeds = (const float*)d_in[0];
  // d_in[1] = attention_mask (causal; applied analytically)
  const float* Wq = (const float*)d_in[2];
  const float* Wk = (const float*)d_in[3];
  const float* Wv = (const float*)d_in[4];
  const float* Wo = (const float*)d_in[5];
  const float* bo = (const float*)d_in[6];
  const float* W1 = (const float*)d_in[7];
  const float* W2 = (const float*)d_in[8];
  const float* b2 = (const float*)d_in[9];
  const float* g1 = (const float*)d_in[10];
  const float* g2 = (const float*)d_in[11];
  const float* gf = (const float*)d_in[12];

  const size_t SH = (size_t)SEQ * H;
  char* ws = (char*)d_ws;
  size_t off = 0;
  float* x  = (float*)(ws + off); off += SH * 4;                 // residual stream
  bf16*  h  = (bf16*)(ws + off);  off += SH * 2;                 // LN output
  bf16*  qb_ = (bf16*)(ws + off); off += SH * 2;
  bf16*  kb_ = (bf16*)(ws + off); off += SH * 2;
  bf16*  vb_ = (bf16*)(ws + off); off += SH * 2;
  bf16*  vtb = (bf16*)(ws + off); off += SH * 2;                 // V^T [H][S]
  bf16*  cx  = (bf16*)(ws + off); off += SH * 2;
  bf16*  t   = (bf16*)(ws + off); off += (size_t)SEQ * IM * 2;   // MLP inter
  bf16*  wt  = (bf16*)(ws + off); off += (size_t)H * IM * 2;     // bf16 Wt [N][K]
  // total ~130 MB

  hipMemcpyAsync(x, embeds, SH * 4, hipMemcpyDeviceToDevice, stream);

  dim3 blk(256);
  dim3 gHH(H / BN, SEQ / BM);     // (16,16)
  dim3 gHI(IM / BN, SEQ / BM);    // (64,16)
  dim3 gAT(NH, SEQ / BQ);         // (32,16)
  dim3 tHH(H / 32, H / 32);       // transpose grids: (C/32, R/32)
  dim3 tHI(IM / 32, H / 32);
  dim3 tIH(H / 32, IM / 32);
  dim3 tSH(H / 32, SEQ / 32);

  for (int l = 0; l < NLAYERS; ++l) {
    const float* wq = Wq + (size_t)l * H * H;
    const float* wk = Wk + (size_t)l * H * H;
    const float* wv = Wv + (size_t)l * H * H;
    const float* wo = Wo + (size_t)l * H * H;
    const float* w1 = W1 + (size_t)l * H * IM;
    const float* w2 = W2 + (size_t)l * IM * H;

    // --- attention block ---
    ln_kernel<bf16><<<SEQ, blk, 0, stream>>>(x, g1 + (size_t)l * H, h);
    transp_kernel<float><<<tHH, blk, 0, stream>>>(wq, wt, H, H);
    gemm_bf16_kernel<1, 0, 0><<<gHH, blk, 0, stream>>>(h, wt, qb_, nullptr, nullptr, SEQ, H, H);
    transp_kernel<float><<<tHH, blk, 0, stream>>>(wk, wt, H, H);
    gemm_bf16_kernel<1, 0, 0><<<gHH, blk, 0, stream>>>(h, wt, kb_, nullptr, nullptr, SEQ, H, H);
    transp_kernel<float><<<tHH, blk, 0, stream>>>(wv, wt, H, H);
    gemm_bf16_kernel<1, 0, 0><<<gHH, blk, 0, stream>>>(h, wt, vb_, nullptr, nullptr, SEQ, H, H);
    transp_kernel<bf16><<<tSH, blk, 0, stream>>>(vb_, vtb, SEQ, H);
    attn_kernel<<<gAT, blk, 0, stream>>>(qb_, kb_, vtb, cx);
    transp_kernel<float><<<tHH, blk, 0, stream>>>(wo, wt, H, H);
    gemm_bf16_kernel<0, 1, 0><<<gHH, blk, 0, stream>>>(cx, wt, x, bo + (size_t)l * H, x, SEQ, H, H);

    // --- MLP block ---
    ln_kernel<bf16><<<SEQ, blk, 0, stream>>>(x, g2 + (size_t)l * H, h);
    transp_kernel<float><<<tHI, blk, 0, stream>>>(w1, wt, H, IM);
    gemm_bf16_kernel<1, 0, 1><<<gHI, blk, 0, stream>>>(h, wt, t, nullptr, nullptr, SEQ, IM, H);
    transp_kernel<float><<<tIH, blk, 0, stream>>>(w2, wt, IM, H);
    gemm_bf16_kernel<0, 1, 0><<<gHH, blk, 0, stream>>>(t, wt, x, b2 + (size_t)l * H, x, SEQ, H, IM);
  }
  ln_kernel<float><<<SEQ, blk, 0, stream>>>(x, gf, (float*)d_out);
}
